// HNPLinear_52055003627835
// MI455X (gfx1250) — matrix operand order repaired and checked by reading, not compiled, not run on hardware
//
#include <hip/hip_runtime.h>

typedef __attribute__((ext_vector_type(16))) __bf16 v16bf;
typedef __attribute__((ext_vector_type(8)))  __bf16 v8bf;
typedef __attribute__((ext_vector_type(8)))  float  v8f;

// ---------------------------------------------------------------------------
// block-wide reduction over 128 threads
// ---------------------------------------------------------------------------
static __device__ __forceinline__ float block_reduce_128(float* red, int t, float v) {
  red[t] = v;
  __syncthreads();
  for (int s = 64; s > 0; s >>= 1) {
    if (t < s) red[t] += red[t + s];
    __syncthreads();
  }
  float r = red[0];
  __syncthreads();
  return r;
}

// ---------------------------------------------------------------------------
// K1: per-(b,c) statistics.  grid = B*C = 2048 blocks, 128 threads
// ---------------------------------------------------------------------------
__global__ __launch_bounds__(128) void stats_kernel(
    const float* __restrict__ w0, const float* __restrict__ w1, const float* __restrict__ w2,
    const float* __restrict__ b0, const float* __restrict__ b1, const float* __restrict__ b2,
    float* __restrict__ s_cm0, float* __restrict__ s_rm1, float* __restrict__ s_cm1,
    float* __restrict__ s_rm2, float* __restrict__ s_rm0, float* __restrict__ s_cm2,
    float* __restrict__ s_sc)
{
  __shared__ float red[128];
  const int bc = blockIdx.x;
  const int t  = threadIdx.x;
  const float* W1 = w1 + (size_t)bc * (128 * 128);
  const float* W0 = w0 + (size_t)bc * (128 * 16);
  const float* W2 = w2 + (size_t)bc * (10 * 128);

  // w1 column sums over d2 -> rm1 ; grand total -> rcm1
  float cs = 0.f;
  #pragma unroll 4
  for (int r = 0; r < 128; ++r) cs += W1[r * 128 + t];
  s_rm1[(size_t)bc * 128 + t] = cs * (1.f / 128.f);
  float tot = block_reduce_128(red, t, cs);
  if (t == 0) s_sc[bc * 8 + 1] = tot * (1.f / 16384.f);

  // w1 row sums over d1 -> cm1
  {
    const float4* row = (const float4*)(W1 + (size_t)t * 128);
    float rs = 0.f;
    #pragma unroll 8
    for (int q = 0; q < 32; ++q) { float4 v4 = row[q]; rs += v4.x + v4.y + v4.z + v4.w; }
    s_cm1[(size_t)bc * 128 + t] = rs * (1.f / 128.f);
  }

  // w0 row sums over NI -> cm0 ; grand total -> rcm0
  {
    const float4* row = (const float4*)(W0 + (size_t)t * 16);
    float rs = 0.f;
    #pragma unroll
    for (int q = 0; q < 4; ++q) { float4 v4 = row[q]; rs += v4.x + v4.y + v4.z + v4.w; }
    s_cm0[(size_t)bc * 128 + t] = rs * (1.f / 16.f);
    float t0 = block_reduce_128(red, t, rs);
    if (t == 0) s_sc[bc * 8 + 0] = t0 * (1.f / 2048.f);
  }

  // w0 column sums over D1 -> rm0
  if (t < 16) {
    float s = 0.f;
    for (int r = 0; r < 128; ++r) s += W0[r * 16 + t];
    s_rm0[bc * 16 + t] = s * (1.f / 128.f);
  }

  // w2 column sums over NO -> rm2 ; grand total -> rcm2
  {
    float c2 = 0.f;
    #pragma unroll
    for (int r = 0; r < 10; ++r) c2 += W2[r * 128 + t];
    s_rm2[(size_t)bc * 128 + t] = c2 * (1.f / 10.f);
    float t2 = block_reduce_128(red, t, c2);
    if (t == 0) s_sc[bc * 8 + 2] = t2 * (1.f / 1280.f);
  }

  // w2 row sums over D2 -> cm2
  if (t < 10) {
    const float4* row = (const float4*)(W2 + (size_t)t * 128);
    float rs = 0.f;
    for (int q = 0; q < 32; ++q) { float4 v4 = row[q]; rs += v4.x + v4.y + v4.z + v4.w; }
    s_cm2[bc * 16 + t] = rs * (1.f / 128.f);
  }

  // bias means
  {
    float t3 = block_reduce_128(red, t, b0[(size_t)bc * 128 + t]);
    if (t == 0) s_sc[bc * 8 + 3] = t3 * (1.f / 128.f);
    float t4 = block_reduce_128(red, t, b1[(size_t)bc * 128 + t]);
    if (t == 0) s_sc[bc * 8 + 4] = t4 * (1.f / 128.f);
    if (t == 0) {
      float s = 0.f;
      for (int m = 0; m < 10; ++m) s += b2[bc * 10 + m];
      s_sc[bc * 8 + 5] = s * 0.1f;
    }
  }
}

// ---------------------------------------------------------------------------
// K2: build rc_inp [B,336].  grid = B, 352 threads
// order: rcm0,rcm1,rcm2,bm0,bm1,bm2 (8 each), rm0 (128), cm2 (80), b2 (80)
// ---------------------------------------------------------------------------
__global__ __launch_bounds__(352) void rc_build_kernel(
    const float* __restrict__ s_sc, const float* __restrict__ s_rm0,
    const float* __restrict__ s_cm2, const float* __restrict__ b2v,
    float* __restrict__ rc_inp)
{
  const int b = blockIdx.x, t = threadIdx.x;
  if (t >= 336) return;
  float v;
  if (t < 48) {
    int seg = t >> 3, c = t & 7;
    v = s_sc[(b * 8 + c) * 8 + seg];
  } else if (t < 176) {
    int idx = t - 48; int c = idx >> 4, i = idx & 15;
    v = s_rm0[(b * 8 + c) * 16 + i];
  } else if (t < 256) {
    int idx = t - 176; int c = idx / 10, o = idx % 10;
    v = s_cm2[(b * 8 + c) * 16 + o];
  } else {
    v = b2v[b * 80 + (t - 256)];
  }
  rc_inp[b * 336 + t] = v;
}

// ---------------------------------------------------------------------------
// K3: all rc-linears.  rcout[b][j]:
//  j 0..255 rbd, 256..271 bias0rc, 272..287 l1rc, 288..303 bias1rc,
//  304..463 cbd, 464..623 finrc (also written straight to u2 output)
// ---------------------------------------------------------------------------
__global__ __launch_bounds__(640) void rc_linear_kernel(
    const float* __restrict__ rc_inp,
    const float* __restrict__ W_rbd,     const float* __restrict__ b_rbd,
    const float* __restrict__ W_bias0rc, const float* __restrict__ b_bias0rc,
    const float* __restrict__ W_l1rc,    const float* __restrict__ b_l1rc,
    const float* __restrict__ W_bias1rc, const float* __restrict__ b_bias1rc,
    const float* __restrict__ W_cbd,     const float* __restrict__ b_cbd,
    const float* __restrict__ W_finrc,   const float* __restrict__ b_finrc,
    float* __restrict__ rcout, float* __restrict__ out_u2)
{
  __shared__ float sh[336];
  const int b = blockIdx.x, t = threadIdx.x;
  if (t < 336) sh[t] = rc_inp[b * 336 + t];
  __syncthreads();
  if (t >= 624) return;
  const float* W; float bias; int row;
  if (t < 256)      { W = W_rbd;     row = t;       bias = b_rbd[row]; }
  else if (t < 272) { W = W_bias0rc; row = t - 256; bias = b_bias0rc[row]; }
  else if (t < 288) { W = W_l1rc;    row = t - 272; bias = b_l1rc[row]; }
  else if (t < 304) { W = W_bias1rc; row = t - 288; bias = b_bias1rc[row]; }
  else if (t < 464) { W = W_cbd;     row = t - 304; bias = b_cbd[row]; }
  else              { W = W_finrc;   row = t - 464; bias = b_finrc[row]; }
  const float* wr = W + (size_t)row * 336;
  float acc = bias;
  #pragma unroll 4
  for (int k = 0; k < 336; ++k) acc += wr[k] * sh[k];
  rcout[b * 624 + t] = acc;
  if (t >= 464) out_u2[b * 160 + (t - 464)] = acc;
}

// ---------------------------------------------------------------------------
// K4a: pack weight (A) matrices to bf16, row-major [M][Kpad] with zero pad.
// A0 [272][160] = {W_rpt; W_bias0} (K=144), A1r [16][160] = W_l1r (K=152),
// A1c [32][128] = {W_l1c; W_bias1} (K=104), A2 [160][96] = W_cpt (K=88)
// ---------------------------------------------------------------------------
__global__ __launch_bounds__(256) void pack_a_kernel(
    const float* __restrict__ W_rpt, const float* __restrict__ W_bias0,
    const float* __restrict__ W_l1r, const float* __restrict__ W_l1c,
    const float* __restrict__ W_bias1, const float* __restrict__ W_cpt,
    __bf16* __restrict__ A0, __bf16* __restrict__ A1r,
    __bf16* __restrict__ A1c, __bf16* __restrict__ A2)
{
  const int idx0 = blockIdx.x * blockDim.x + threadIdx.x;
  const int stride = gridDim.x * blockDim.x;
  for (int idx = idx0; idx < 272 * 160; idx += stride) {
    int m = idx / 160, k = idx % 160;
    float v = 0.f;
    if (k < 144) v = (m < 256) ? W_rpt[m * 144 + k] : W_bias0[(m - 256) * 144 + k];
    A0[idx] = (__bf16)v;
  }
  for (int idx = idx0; idx < 16 * 160; idx += stride) {
    int m = idx / 160, k = idx % 160;
    A1r[idx] = (__bf16)((k < 152) ? W_l1r[m * 152 + k] : 0.f);
  }
  for (int idx = idx0; idx < 32 * 128; idx += stride) {
    int m = idx / 128, k = idx % 128;
    float v = 0.f;
    if (k < 104) v = (m < 16) ? W_l1c[m * 104 + k] : W_bias1[(m - 16) * 104 + k];
    A1c[idx] = (__bf16)v;
  }
  for (int idx = idx0; idx < 160 * 96; idx += stride) {
    int m = idx / 96, k = idx % 96;
    A2[idx] = (__bf16)((k < 88) ? W_cpt[m * 88 + k] : 0.f);
  }
}

// ---------------------------------------------------------------------------
// K4b: pack per-batch activation (B) matrices, transposed K-major bf16:
// X_T[b][n][k] = x[b][k][n].  grid = B, 160 threads (thread = k)
// ---------------------------------------------------------------------------
__global__ __launch_bounds__(160) void pack_b_kernel(
    const float* __restrict__ w0, const float* __restrict__ w2,
    const float* __restrict__ b0v, const float* __restrict__ b1v,
    const float* __restrict__ s_rm1, const float* __restrict__ s_cm0,
    const float* __restrict__ s_cm1, const float* __restrict__ s_rm2,
    __bf16* __restrict__ BrptT, __bf16* __restrict__ BrowT,
    __bf16* __restrict__ BcolT, __bf16* __restrict__ BcptT)
{
  const int b = blockIdx.x;
  const int k = threadIdx.x;  // 0..159
  // rpt = [w0t(128), rm1(8), b0(8)]  -> BrptT [128][160]
  for (int n = 0; n < 128; ++n) {
    float v = 0.f;
    if (k < 128)      { int c = k >> 4, i = k & 15; v = w0[(((size_t)b * 8 + c) * 128 + n) * 16 + i]; }
    else if (k < 136) v = s_rm1[((size_t)b * 8 + (k - 128)) * 128 + n];
    else if (k < 144) v = b0v[((size_t)b * 8 + (k - 136)) * 128 + n];
    BrptT[((size_t)b * 128 + n) * 160 + k] = (__bf16)v;
  }
  // rowb = [rm1(8), cm0(8), b0(8), w0t(128)] -> BrowT [128][160]
  for (int n = 0; n < 128; ++n) {
    float v = 0.f;
    if (k < 8)        v = s_rm1[((size_t)b * 8 + k) * 128 + n];
    else if (k < 16)  v = s_cm0[((size_t)b * 8 + (k - 8)) * 128 + n];
    else if (k < 24)  v = b0v[((size_t)b * 8 + (k - 16)) * 128 + n];
    else if (k < 152) { int r = k - 24; int c = r >> 4, i = r & 15; v = w0[(((size_t)b * 8 + c) * 128 + n) * 16 + i]; }
    BrowT[((size_t)b * 128 + n) * 160 + k] = (__bf16)v;
  }
  // colb = [cm1(8), b1(8), rm2(8), w2f(80)] -> BcolT [128][128]
  if (k < 128) {
    for (int n = 0; n < 128; ++n) {
      float v = 0.f;
      if (k < 8)        v = s_cm1[((size_t)b * 8 + k) * 128 + n];
      else if (k < 16)  v = b1v[((size_t)b * 8 + (k - 8)) * 128 + n];
      else if (k < 24)  v = s_rm2[((size_t)b * 8 + (k - 16)) * 128 + n];
      else if (k < 104) { int r = k - 24; int c = r / 10, o = r % 10; v = w2[(((size_t)b * 8 + c) * 10 + o) * 128 + n]; }
      BcolT[((size_t)b * 128 + n) * 128 + k] = (__bf16)v;
    }
  }
  // cpt = [w2f(80), cm1(8)] -> BcptT [128][96]
  if (k < 96) {
    for (int n = 0; n < 128; ++n) {
      float v = 0.f;
      if (k < 80)      { int c = k / 10, o = k % 10; v = w2[(((size_t)b * 8 + c) * 10 + o) * 128 + n]; }
      else if (k < 88) v = s_cm1[((size_t)b * 8 + (k - 80)) * 128 + n];
      BcptT[((size_t)b * 128 + n) * 96 + k] = (__bf16)v;
    }
  }
}

// ---------------------------------------------------------------------------
// WMMA batched GEMM: Out[b] = A @ x[b]   (A: [Mtiles*16][K], x_T: [128][K])
// grid = (B, ceil(Mtiles/4)), block = 32..128 (each wave owns one M-tile)
// ---------------------------------------------------------------------------
template <int KSTEPS>
__global__ __launch_bounds__(128) void wmma_gemm(
    const __bf16* __restrict__ A,   // [Mtiles*16][KSTEPS*32] row-major
    const __bf16* __restrict__ Bm,  // [Bsz][128][KSTEPS*32]  (x transposed)
    float* __restrict__ Out,        // [Bsz][Mtiles*16][128]
    int Mtiles)
{
  const int lane  = threadIdx.x & 31;
  const int wave  = threadIdx.x >> 5;
  const int mtile = blockIdx.y * 4 + wave;
  if (mtile >= Mtiles) return;                 // wave-uniform: EXEC stays full
  const int b    = blockIdx.x;
  const int K    = KSTEPS * 32;
  const int half = lane >> 4;
  const int l15  = lane & 15;

  // A fragments: lane holds row m = l15; element e -> k = ks*32 + half*16 + e
  const __bf16* arow = A + (size_t)(mtile * 16 + l15) * K;
  v16bf a[KSTEPS];
  #pragma unroll
  for (int ks = 0; ks < KSTEPS; ++ks) {
    const __bf16* p = arow + ks * 32 + half * 16;
    v8bf lo = *(const v8bf*)(p);
    v8bf hi = *(const v8bf*)(p + 8);
    #pragma unroll
    for (int e = 0; e < 8; ++e) { a[ks][e] = lo[e]; a[ks][e + 8] = hi[e]; }
  }

  const __bf16* bbase = Bm + (size_t)b * 128 * K;
  float* obase = Out + ((size_t)b * Mtiles * 16 + (size_t)mtile * 16) * 128;

  #pragma unroll 1
  for (int nt = 0; nt < 8; ++nt) {
    // B fragment: lane holds col n = l15, k = ks*32 + half*16 + e (contiguous 16)
    const __bf16* brow = bbase + (size_t)(nt * 16 + l15) * K + half * 16;
    v8f acc = {};
    #pragma unroll
    for (int ks = 0; ks < KSTEPS; ++ks) {
      v16bf bv = *(const v16bf*)(brow + ks * 32);
      acc = __builtin_amdgcn_wmma_f32_16x16x32_bf16(
          false, a[ks], false, bv, (short)0, acc, false, false);
    }
    #pragma unroll
    for (int v = 0; v < 8; ++v) {
      int m = v + half * 8;
      obase[(size_t)m * 128 + nt * 16 + l15] = acc[v];
    }
  }
}

// ---------------------------------------------------------------------------
// E1: zw0 (transpose-scatter) + u0.  grid (B,16), 256 threads
// ---------------------------------------------------------------------------
__global__ __launch_bounds__(256) void epilogue0_kernel(
    const float* __restrict__ Z0, const float* __restrict__ rcout,
    const float* __restrict__ b_rpt, const float* __restrict__ b_bias0,
    float* __restrict__ out_zw0, float* __restrict__ out_u0)
{
  const int b = blockIdx.x, o = blockIdx.y, t = threadIdx.x;
  const int i = t & 15;
  const int n0 = t >> 4;
  const int j = o * 16 + i;
  const float add = b_rpt[j] + rcout[b * 624 + j];
  #pragma unroll
  for (int rep = 0; rep < 8; ++rep) {
    int n = n0 + rep * 16;
    float v = Z0[((size_t)b * 272 + j) * 128 + n] + add;
    out_zw0[(((size_t)b * 16 + o) * 128 + n) * 16 + i] = v;
  }
  if (t < 128) {
    float v = Z0[((size_t)b * 272 + 256 + o) * 128 + t] + b_bias0[o] + rcout[b * 624 + 256 + o];
    out_u0[((size_t)b * 16 + o) * 128 + t] = v;
  }
}

// ---------------------------------------------------------------------------
// E2: fused zw1 = W_l1 . w1 + broadcasts (memory-bound main pass)
// grid = B*128 (one (b,h) per block), 128 threads (w)
// ---------------------------------------------------------------------------
__global__ __launch_bounds__(128) void zw1_kernel(
    const float* __restrict__ w1, const float* __restrict__ W_l1,
    const float* __restrict__ b_l1, const float* __restrict__ b_l1r,
    const float* __restrict__ b_l1c,
    const float* __restrict__ Z1r, const float* __restrict__ Z1cb,
    const float* __restrict__ rcout, float* __restrict__ out_zw1)
{
  __shared__ float wl1s[128];
  __shared__ float konst[16];
  const int h = blockIdx.x & 127;
  const int b = blockIdx.x >> 7;
  const int t = threadIdx.x;
  wl1s[t] = W_l1[t];
  if (t < 16)
    konst[t] = rcout[b * 624 + 272 + t] + b_l1[t] + b_l1r[t] + b_l1c[t]
             + Z1cb[((size_t)b * 32 + t) * 128 + h];
  __syncthreads();
  const int w = t;
  float v[8];
  #pragma unroll
  for (int i = 0; i < 8; ++i)
    v[i] = w1[(((size_t)b * 8 + i) * 128 + h) * 128 + w];
  #pragma unroll 4
  for (int o = 0; o < 16; ++o) {
    float acc = konst[o] + Z1r[((size_t)b * 16 + o) * 128 + w];
    #pragma unroll
    for (int i = 0; i < 8; ++i) acc += wl1s[o * 8 + i] * v[i];
    out_zw1[(((size_t)b * 16 + o) * 128 + h) * 128 + w] = acc;
  }
}

// ---------------------------------------------------------------------------
// E3: u1.  grid = B, 128 threads (h)
// ---------------------------------------------------------------------------
__global__ __launch_bounds__(128) void u1_kernel(
    const float* __restrict__ Z1cb, const float* __restrict__ rcout,
    const float* __restrict__ b_bias1, float* __restrict__ out_u1)
{
  const int b = blockIdx.x, h = threadIdx.x;
  #pragma unroll
  for (int o = 0; o < 16; ++o)
    out_u1[((size_t)b * 16 + o) * 128 + h] =
        Z1cb[((size_t)b * 32 + 16 + o) * 128 + h] + b_bias1[o] + rcout[b * 624 + 288 + o];
}

// ---------------------------------------------------------------------------
// E4: zw2.  grid = B, 128 threads (n)
// ---------------------------------------------------------------------------
__global__ __launch_bounds__(128) void zw2_kernel(
    const float* __restrict__ Z2, const float* __restrict__ rcout,
    const float* __restrict__ b_cpt, float* __restrict__ out_zw2)
{
  const int b = blockIdx.x, n = threadIdx.x;
  for (int r = 0; r < 160; ++r) {
    out_zw2[((size_t)b * 160 + r) * 128 + n] =
        Z2[((size_t)b * 160 + r) * 128 + n] + b_cpt[r] + rcout[b * 624 + 304 + r];
  }
}

// ---------------------------------------------------------------------------
extern "C" void kernel_launch(void* const* d_in, const int* in_sizes, int n_in,
                              void* d_out, int out_size, void* d_ws, size_t ws_size,
                              hipStream_t stream) {
  (void)in_sizes; (void)n_in; (void)out_size; (void)ws_size;
  const float* w0 = (const float*)d_in[0];
  const float* w1 = (const float*)d_in[1];
  const float* w2 = (const float*)d_in[2];
  const float* b0 = (const float*)d_in[3];
  const float* b1 = (const float*)d_in[4];
  const float* b2 = (const float*)d_in[5];
  const float* W_rpt = (const float*)d_in[6];
  const float* b_rpt = (const float*)d_in[7];
  const float* W_rbd = (const float*)d_in[8];
  const float* b_rbd = (const float*)d_in[9];
  const float* W_bias0 = (const float*)d_in[10];
  const float* b_bias0 = (const float*)d_in[11];
  const float* W_bias0rc = (const float*)d_in[12];
  const float* b_bias0rc = (const float*)d_in[13];
  const float* W_l1 = (const float*)d_in[14];
  const float* b_l1 = (const float*)d_in[15];
  const float* W_l1rc = (const float*)d_in[16];
  const float* b_l1rc = (const float*)d_in[17];
  const float* W_l1r = (const float*)d_in[18];
  const float* b_l1r = (const float*)d_in[19];
  const float* W_l1c = (const float*)d_in[20];
  const float* b_l1c = (const float*)d_in[21];
  const float* W_bias1 = (const float*)d_in[22];
  const float* b_bias1 = (const float*)d_in[23];
  const float* W_bias1rc = (const float*)d_in[24];
  const float* b_bias1rc = (const float*)d_in[25];
  const float* W_cpt = (const float*)d_in[26];
  const float* b_cpt = (const float*)d_in[27];
  const float* W_cbd = (const float*)d_in[28];
  const float* b_cbd = (const float*)d_in[29];
  const float* W_finrc = (const float*)d_in[30];
  const float* b_finrc = (const float*)d_in[31];

  float* out = (float*)d_out;
  float* out_zw0 = out;                 // [256,16,128,16]
  float* out_zw1 = out + 8388608;       // [256,16,128,128]
  float* out_zw2 = out + 75497472;      // [256,16,10,128]
  float* out_u0  = out + 80740352;      // [256,16,128]
  float* out_u1  = out + 81264640;      // [256,16,128]
  float* out_u2  = out + 81788928;      // [256,16,10]

  char* ws = (char*)d_ws;
  size_t off = 0;
  auto carve = [&](size_t bytes) -> char* { char* p = ws + off; off += bytes; return p; };
  float* s_cm0 = (float*)carve((size_t)2048 * 128 * 4);
  float* s_rm1 = (float*)carve((size_t)2048 * 128 * 4);
  float* s_cm1 = (float*)carve((size_t)2048 * 128 * 4);
  float* s_rm2 = (float*)carve((size_t)2048 * 128 * 4);
  float* s_rm0 = (float*)carve((size_t)2048 * 16 * 4);
  float* s_cm2 = (float*)carve((size_t)2048 * 16 * 4);
  float* s_sc  = (float*)carve((size_t)2048 * 8 * 4);
  float* rc_inp = (float*)carve((size_t)256 * 336 * 4);
  float* rcout  = (float*)carve((size_t)256 * 624 * 4);
  float* Z0   = (float*)carve((size_t)256 * 272 * 128 * 4);
  float* Z1r  = (float*)carve((size_t)256 * 16 * 128 * 4);
  float* Z1cb = (float*)carve((size_t)256 * 32 * 128 * 4);
  float* Z2   = (float*)carve((size_t)256 * 160 * 128 * 4);
  __bf16* A0   = (__bf16*)carve((size_t)272 * 160 * 2);
  __bf16* A1r  = (__bf16*)carve((size_t)16 * 160 * 2);
  __bf16* A1c  = (__bf16*)carve((size_t)32 * 128 * 2);
  __bf16* A2m  = (__bf16*)carve((size_t)160 * 96 * 2);
  __bf16* BrptT = (__bf16*)carve((size_t)256 * 128 * 160 * 2);
  __bf16* BrowT = (__bf16*)carve((size_t)256 * 128 * 160 * 2);
  __bf16* BcolT = (__bf16*)carve((size_t)256 * 128 * 128 * 2);
  __bf16* BcptT = (__bf16*)carve((size_t)256 * 128 * 96 * 2);

  stats_kernel<<<2048, 128, 0, stream>>>(w0, w1, w2, b0, b1, b2,
                                         s_cm0, s_rm1, s_cm1, s_rm2, s_rm0, s_cm2, s_sc);
  rc_build_kernel<<<256, 352, 0, stream>>>(s_sc, s_rm0, s_cm2, b2, rc_inp);
  rc_linear_kernel<<<256, 640, 0, stream>>>(rc_inp, W_rbd, b_rbd, W_bias0rc, b_bias0rc,
                                            W_l1rc, b_l1rc, W_bias1rc, b_bias1rc,
                                            W_cbd, b_cbd, W_finrc, b_finrc, rcout, out_u2);
  pack_a_kernel<<<32, 256, 0, stream>>>(W_rpt, W_bias0, W_l1r, W_l1c, W_bias1, W_cpt,
                                        A0, A1r, A1c, A2m);
  pack_b_kernel<<<256, 160, 0, stream>>>(w0, w2, b0, b1, s_rm1, s_cm0, s_cm1, s_rm2,
                                         BrptT, BrowT, BcolT, BcptT);
  wmma_gemm<5><<<dim3(256, 5), 128, 0, stream>>>(A0, BrptT, Z0, 17);
  wmma_gemm<5><<<dim3(256, 1), 32, 0, stream>>>(A1r, BrowT, Z1r, 1);
  wmma_gemm<4><<<dim3(256, 1), 64, 0, stream>>>(A1c, BcolT, Z1cb, 2);
  wmma_gemm<3><<<dim3(256, 3), 128, 0, stream>>>(A2m, BcptT, Z2, 10);
  epilogue0_kernel<<<dim3(256, 16), 256, 0, stream>>>(Z0, rcout, b_rpt, b_bias0, out_zw0, out_u0);
  zw1_kernel<<<256 * 128, 128, 0, stream>>>(w1, W_l1, b_l1, b_l1r, b_l1c, Z1r, Z1cb, rcout, out_zw1);
  u1_kernel<<<256, 128, 0, stream>>>(Z1cb, rcout, b_bias1, out_u1);
  zw2_kernel<<<256, 128, 0, stream>>>(Z2, rcout, b_cpt, out_zw2);
}
